// HierCDF_18116172054653
// MI455X (gfx1250) — matrix operand order repaired and checked by reading, not compiled, not run on hardware
//
#include <hip/hip_runtime.h>
#include <hip/hip_bf16.h>

typedef __attribute__((ext_vector_type(16))) _Float16 v16h;
typedef __attribute__((ext_vector_type(8)))  _Float16 v8h;
typedef __attribute__((ext_vector_type(8)))  float    v8f;

#define N_KNOW 128
#define N_EDGE 253
#define HIDDEN 512
#define HID2   256
#define BATCH  16384

__device__ __forceinline__ float sigf(float x) {
    return 1.0f / (1.0f + __expf(-x));
}

// ---------------------------------------------------------------------------
// Kernel 0: one-shot f32 -> f16 weight conversion (weights are tiny and
// reused by 1024 M-tiles; converting once removes all v_cvt_pk_f16_f32 from
// the GEMM inner loops and halves B-side load bytes).
// ---------------------------------------------------------------------------
__global__ __launch_bounds__(256) void cvt_weights_kernel(
    const float* __restrict__ src, _Float16* __restrict__ dst, int n)
{
    int i = blockIdx.x * blockDim.x + threadIdx.x;
    if (i < n) dst[i] = (_Float16)src[i];
}

// ---------------------------------------------------------------------------
// Kernel 1: per-row posterior chain + build f16 A-matrices for the GEMMs.
// One thread per batch row; serial 128-step recurrence (only depends on the
// previous two columns), gathers user rows of priori/condi_p/condi_n.
// ---------------------------------------------------------------------------
__global__ __launch_bounds__(256) void posterior_kernel(
    const int* __restrict__ uid, const int* __restrict__ iid,
    const float* __restrict__ iknow, const float* __restrict__ priori,
    const float* __restrict__ cp, const float* __restrict__ cn,
    const float* __restrict__ idw, const float* __restrict__ idiscw,
    _Float16* __restrict__ userA, _Float16* __restrict__ itemA,
    float* __restrict__ disc)
{
    int b = blockIdx.x * blockDim.x + threadIdx.x;
    if (b >= BATCH) return;
    int u  = uid[b];
    int it = iid[b];

    const float* prow  = priori + (size_t)u * N_KNOW;
    const float* cprow = cp     + (size_t)u * N_EDGE;
    const float* cnrow = cn     + (size_t)u * N_EDGE;
    const float* kn    = iknow  + (size_t)b * N_KNOW;
    const float* dw    = idw    + (size_t)it * N_KNOW;
    _Float16* ua = userA + (size_t)b * N_KNOW;
    _Float16* ia = itemA + (size_t)b * N_KNOW;

    // k = 0 : root node, no predecessors
    float ckm2 = sigf(prow[0]);
    ua[0] = (_Float16)(ckm2 * kn[0]);

    // k = 1 : single predecessor (edge 0), exponent 1
    float c0 = ckm2;
    float ckm1 = sigf(cprow[0]) * c0 + sigf(cnrow[0]) * (1.0f - c0);
    ua[1] = (_Float16)(ckm1 * kn[1]);

    // k >= 2 : two predecessors k-2 (edge 2k-3) and k-1 (edge 2k-2), sqrt weights
    #pragma unroll 4
    for (int k = 2; k < N_KNOW; ++k) {
        int e0 = 2 * k - 3;
        int e1 = 2 * k - 2;
        float cp0 = __fsqrt_rn(sigf(cprow[e0]));
        float cn0 = __fsqrt_rn(sigf(cnrow[e0]));
        float cp1 = __fsqrt_rn(sigf(cprow[e1]));
        float cn1 = __fsqrt_rn(sigf(cnrow[e1]));
        float t0 = cp0 * ckm2 + cn0 * (1.0f - ckm2);
        float t1 = cp1 * ckm1 + cn1 * (1.0f - ckm1);
        float ck = t0 * t1;
        ua[k] = (_Float16)(ck * kn[k]);
        ckm2 = ckm1;
        ckm1 = ck;
    }

    // item side: sigmoid(item_diff) * item_know
    #pragma unroll 8
    for (int k = 0; k < N_KNOW; ++k) {
        ia[k] = (_Float16)(sigf(dw[k]) * kn[k]);
    }
    disc[b] = sigf(idiscw[it]);
}

// ---------------------------------------------------------------------------
// WMMA fragment helpers (wave32; this code's contiguous k order, same for A and B):
//  A (16x32 f16): lane = M (lane&15), khalf = lane>>4.
//    elem i -> K = kbase + khalf*16 + i            (16 contiguous halfs)
//  B (32x16 f16): lane = N (lane&15), khalf = lane>>4.
//    elem i -> K = kbase + khalf*16 + i            (16 contiguous halfs)
//  C/D f32: lane = N (lane&15); elem j -> M = j + 8*(lane>>4)
// ---------------------------------------------------------------------------
__device__ __forceinline__ v16h load_a_frag(const _Float16* __restrict__ row,
                                            int kbase, int khalf)
{
    v8h lo = *(const v8h*)(row + kbase + khalf * 16);
    v8h hi = *(const v8h*)(row + kbase + khalf * 16 + 8);
    v16h a;
    #pragma unroll
    for (int i = 0; i < 8; ++i) { a[i] = lo[i]; a[i + 8] = hi[i]; }
    return a;
}

__device__ __forceinline__ v16h load_b_frag(const _Float16* __restrict__ wrow,
                                            int kbase)
{
    v8h lo = *(const v8h*)(wrow + kbase);
    v8h hi = *(const v8h*)(wrow + kbase + 8);
    v16h bfr;
    #pragma unroll
    for (int i = 0; i < 8; ++i) { bfr[i] = lo[i]; bfr[i + 8] = hi[i]; }
    return bfr;
}

// ---------------------------------------------------------------------------
// Kernel 2: dual fused GEMM (16384x128 @ 128x512) for user & item factors.
// One wave per 16x16 output tile, two accumulators sharing the A-row address
// pattern. Epilogue: x = (tanh(u+bu) - sigmoid(i+bi)) * disc, stored f16.
// Inner loop: 8 x b128 loads + 2 WMMA per K-step, no converts.
// ---------------------------------------------------------------------------
__global__ __launch_bounds__(128) void factor_gemm_kernel(
    const _Float16* __restrict__ userA, const _Float16* __restrict__ itemA,
    const _Float16* __restrict__ ucwh, const float* __restrict__ ucb,
    const _Float16* __restrict__ icwh, const float* __restrict__ icb,
    const float* __restrict__ disc, _Float16* __restrict__ X)
{
    const int NT = HIDDEN / 16;                 // 32 N-tiles
    int wave = threadIdx.x >> 5;
    int lane = threadIdx.x & 31;
    int tile = blockIdx.x * 4 + wave;
    int mt = tile / NT;
    int nt = tile - mt * NT;
    int m0 = mt * 16;
    int n0 = nt * 16;

    int l15   = lane & 15;
    int khalf = lane >> 4;
    int n     = n0 + l15;

    const _Float16* ua = userA + (size_t)(m0 + l15) * N_KNOW;
    const _Float16* ia = itemA + (size_t)(m0 + l15) * N_KNOW;
    const _Float16* wu = ucwh + (size_t)n * N_KNOW + khalf * 16;
    const _Float16* wi = icwh + (size_t)n * N_KNOW + khalf * 16;

    v8f accU = {};
    v8f accI = {};
    #pragma unroll
    for (int kb = 0; kb < N_KNOW / 32; ++kb) {   // 4 K-steps
        int kbase = kb * 32;
        v16h aU = load_a_frag(ua, kbase, khalf);
        v16h aI = load_a_frag(ia, kbase, khalf);
        v16h bU = load_b_frag(wu, kbase);
        v16h bI = load_b_frag(wi, kbase);
        accU = __builtin_amdgcn_wmma_f32_16x16x32_f16(
            false, aU, false, bU, (short)0, accU, false, false);
        accI = __builtin_amdgcn_wmma_f32_16x16x32_f16(
            false, aI, false, bI, (short)0, accI, false, false);
    }

    float bu = ucb[n];
    float bi = icb[n];
    #pragma unroll
    for (int j = 0; j < 8; ++j) {
        int m = m0 + j + 8 * khalf;
        float uf  = tanhf(accU[j] + bu);
        float itf = sigf(accI[j] + bi);
        float xv  = (uf - itf) * disc[m];
        X[(size_t)m * HIDDEN + n] = (_Float16)xv;
    }
}

// ---------------------------------------------------------------------------
// Kernel 3: (16384x512 f16) @ (512x256) -> sigmoid -> dot with c2_w -> sigmoid.
// One block per 16-row tile; 16 waves each own one 16-col tile (covers all 256
// hidden2 columns). h never touches memory: sigmoid + scale by c2_w in regs,
// shfl_xor tree-reduce across the 16 lanes holding one row, LDS ds_add_f32
// across waves, final sigmoid to out.
// ---------------------------------------------------------------------------
__global__ __launch_bounds__(512) void head_kernel(
    const _Float16* __restrict__ X,
    const _Float16* __restrict__ c1wh, const float* __restrict__ c1b,
    const float* __restrict__ c2w, const float* __restrict__ c2b,
    float* __restrict__ out)
{
    __shared__ float acc[16];
    int wave = threadIdx.x >> 5;
    int lane = threadIdx.x & 31;
    int m0 = blockIdx.x * 16;

    if (threadIdx.x < 16) acc[threadIdx.x] = 0.0f;
    __syncthreads();

    int l15   = lane & 15;
    int khalf = lane >> 4;
    int n     = wave * 16 + l15;                 // 0..255

    const _Float16* xa = X + (size_t)(m0 + l15) * HIDDEN;
    const _Float16* w  = c1wh + (size_t)n * HIDDEN + khalf * 16;

    v8f accD = {};
    #pragma unroll
    for (int kb = 0; kb < HIDDEN / 32; ++kb) {   // 16 K-steps
        int kbase = kb * 32;
        v16h a   = load_a_frag(xa, kbase, khalf);
        v16h bfr = load_b_frag(w, kbase);
        accD = __builtin_amdgcn_wmma_f32_16x16x32_f16(
            false, a, false, bfr, (short)0, accD, false, false);
    }

    float b1  = c1b[n];
    float c2n = c2w[n];
    #pragma unroll
    for (int j = 0; j < 8; ++j) {
        float h = sigf(accD[j] + b1) * c2n;      // sigmoid(x@c1+b1) * c2_w[n]
        h += __shfl_xor(h, 1);
        h += __shfl_xor(h, 2);
        h += __shfl_xor(h, 4);
        h += __shfl_xor(h, 8);                   // sum over 16 columns in wave
        if (l15 == 0) atomicAdd(&acc[j + 8 * khalf], h);
    }
    __syncthreads();

    if (threadIdx.x < 16) {
        out[m0 + threadIdx.x] = sigf(acc[threadIdx.x] + c2b[0]);
    }
}

// ---------------------------------------------------------------------------
extern "C" void kernel_launch(void* const* d_in, const int* in_sizes, int n_in,
                              void* d_out, int out_size, void* d_ws, size_t ws_size,
                              hipStream_t stream)
{
    (void)in_sizes; (void)n_in; (void)out_size; (void)ws_size;
    const int*   uid    = (const int*)d_in[0];
    const int*   iid    = (const int*)d_in[1];
    const float* iknow  = (const float*)d_in[2];
    const float* priori = (const float*)d_in[3];
    const float* cp     = (const float*)d_in[4];
    const float* cn     = (const float*)d_in[5];
    const float* idw    = (const float*)d_in[6];
    const float* idiscw = (const float*)d_in[7];
    const float* ucw    = (const float*)d_in[8];
    const float* ucb    = (const float*)d_in[9];
    const float* icw    = (const float*)d_in[10];
    const float* icb    = (const float*)d_in[11];
    const float* c1w    = (const float*)d_in[12];
    const float* c1b    = (const float*)d_in[13];
    const float* c2w    = (const float*)d_in[14];
    const float* c2b    = (const float*)d_in[15];
    float* out = (float*)d_out;

    // Workspace layout (all regions 256B+ aligned):
    char* ws = (char*)d_ws;
    const size_t MB = 1048576;
    _Float16* userA = (_Float16*)(ws);                    //  4 MiB
    _Float16* itemA = (_Float16*)(ws + 4  * MB);          //  4 MiB
    _Float16* X     = (_Float16*)(ws + 8  * MB);          // 16 MiB
    float*    disc  = (float*)   (ws + 24 * MB);          // 64 KiB
    _Float16* ucwh  = (_Float16*)(ws + 24 * MB + 131072); // 128 KiB
    _Float16* icwh  = (_Float16*)(ws + 24 * MB + 262144); // 128 KiB
    _Float16* c1wh  = (_Float16*)(ws + 24 * MB + 393216); // 256 KiB

    // One-shot weight conversions (f32 -> f16)
    cvt_weights_kernel<<<(HIDDEN * N_KNOW) / 256, 256, 0, stream>>>(
        ucw, ucwh, HIDDEN * N_KNOW);
    cvt_weights_kernel<<<(HIDDEN * N_KNOW) / 256, 256, 0, stream>>>(
        icw, icwh, HIDDEN * N_KNOW);
    cvt_weights_kernel<<<(HID2 * HIDDEN) / 256, 256, 0, stream>>>(
        c1w, c1wh, HID2 * HIDDEN);

    posterior_kernel<<<BATCH / 256, 256, 0, stream>>>(
        uid, iid, iknow, priori, cp, cn, idw, idiscw, userA, itemA, disc);

    // (BATCH/16) * (HIDDEN/16) = 32768 wave-tiles, 4 waves per block
    factor_gemm_kernel<<<(BATCH / 16) * (HIDDEN / 16) / 4, 128, 0, stream>>>(
        userA, itemA, ucwh, ucb, icwh, icb, disc, X);

    head_kernel<<<BATCH / 16, 512, 0, stream>>>(
        X, c1wh, c1b, c2w, c2b, out);
}
